// Encoder_8589935264
// MI455X (gfx1250) — matrix operand order repaired and checked by reading, not compiled, not run on hardware
//
#include <hip/hip_runtime.h>
#include <hip/hip_bf16.h>

// ---------------------------------------------------------------------------
// CDNA5 (gfx1250) wave32 WMMA types
// ---------------------------------------------------------------------------
typedef __attribute__((ext_vector_type(16))) __bf16        v16bf;
typedef __attribute__((ext_vector_type(8)))  float         v8f;
typedef __attribute__((ext_vector_type(8)))  unsigned int  v8u;

// f32 -> bf16 with round-to-nearest-even (bit trick; used only in epilogues /
// conversion passes, never in the GEMM inner loop).
static __device__ __forceinline__ __bf16 f2bf(float f) {
  unsigned u = __builtin_bit_cast(unsigned, f);
  u = u + 0x7FFFu + ((u >> 16) & 1u);
  unsigned short s = (unsigned short)(u >> 16);
  return __builtin_bit_cast(__bf16, s);
}

// ---------------------------------------------------------------------------
// Batched WMMA GEMM over pre-converted bf16 operands (NT form):
//   A: bf16 [M,K] row-major (lda), B: bf16 [N,K] row-major (ldb)
//   requires M%16==0, N%16==0, K%128==0 -> no guards, 4 WMMAs per K-iter.
// Epilogue: + bias(f32) + residual(f32), optional ReLU, writes optional
// f32 C and/or bf16 C (optionally transposed, used to build V^T).
// ---------------------------------------------------------------------------
struct GemmArgs {
  const __bf16* A; const __bf16* B;
  float*  Cf;                   // optional f32 out [M,N] (ldcf)
  __bf16* Cb;                   // optional bf16 out (ldcb)
  const float* bias;            // [N] or nullptr
  const float* residual;        // f32 [M,N] (ldr) or nullptr
  int  M, N, K;
  int  lda, ldb, ldcf, ldcb, ldr;
  long sA, sB, sCf, sCb, sR;    // batch strides (elements)
  int  batch, relu, transOut;   // transOut: Cb written as [N,M]
};

__global__ __launch_bounds__(256) void wmma_gemm_kernel(GemmArgs g) {
  const int lane = threadIdx.x & 31;
  const int wave = threadIdx.x >> 5;
  const int hi   = lane >> 4;     // lane half
  const int lo   = lane & 15;

  const int  mTiles = g.M >> 4;
  const int  nTiles = g.N >> 4;
  const long tilesPerBatch = (long)mTiles * nTiles;
  const long totalTiles    = tilesPerBatch * g.batch;

  for (long tile = (long)blockIdx.x * 8 + wave; tile < totalTiles;
       tile += (long)gridDim.x * 8) {
    const int  b  = (int)(tile / tilesPerBatch);
    const long t  = tile - (long)b * tilesPerBatch;
    const int  mt = (int)(t / nTiles);
    const int  nt = (int)(t % nTiles);
    const int  m0 = mt << 4, n0 = nt << 4;

    // A fragment (ISA 16-bit A 16x32 layout): lane half hi holds K-runs
    //   {kb+8hi..+7} and {kb+16+8hi..+7}  -> two 16B loads per step.
    const __bf16* Arow =
        g.A + (long)b * g.sA + (long)(m0 + lo) * g.lda + (hi << 3);
    // B fragment: same K order as A: lane half hi holds K-runs
    //   {kb+8hi..+7} and {kb+16+8hi..+7}, column n0+lo; B stored [N,K].
    const __bf16* Brow =
        g.B + (long)b * g.sB + (long)(n0 + lo) * g.ldb + (hi << 3);

    v8f acc = {};
    for (int k0 = 0; k0 < g.K; k0 += 128) {
#pragma unroll
      for (int s = 0; s < 4; ++s) {
        const int kb = k0 + 32 * s;
        const uint4 a0 = *(const uint4*)(Arow + kb);
        const uint4 a1 = *(const uint4*)(Arow + kb + 16);
        const uint4 b0 = *(const uint4*)(Brow + kb);
        const uint4 b1 = *(const uint4*)(Brow + kb + 16);
        v8u au, bu;
        au[0] = a0.x; au[1] = a0.y; au[2] = a0.z; au[3] = a0.w;
        au[4] = a1.x; au[5] = a1.y; au[6] = a1.z; au[7] = a1.w;
        bu[0] = b0.x; bu[1] = b0.y; bu[2] = b0.z; bu[3] = b0.w;
        bu[4] = b1.x; bu[5] = b1.y; bu[6] = b1.z; bu[7] = b1.w;
        acc = __builtin_amdgcn_wmma_f32_16x16x32_bf16(
            false, __builtin_bit_cast(v16bf, au),
            false, __builtin_bit_cast(v16bf, bu),
            (short)0, acc, false, false);
      }
    }

    // Epilogue: lane lo = N, vgpr r -> M = m0 + 8*hi + r (padded, unguarded)
    const int   n     = n0 + lo;
    const int   mBase = m0 + (hi << 3);
    const float bv    = g.bias ? g.bias[n] : 0.0f;
    const float* pr = g.residual
        ? g.residual + (long)b * g.sR + (long)mBase * g.ldr + n : nullptr;
    float* pf = g.Cf
        ? g.Cf + (long)b * g.sCf + (long)mBase * g.ldcf + n : nullptr;
    __bf16* pb = nullptr;
    if (g.Cb)
      pb = g.Cb + (long)b * g.sCb +
           (g.transOut ? ((long)n * g.ldcb + mBase)
                       : ((long)mBase * g.ldcb + n));
#pragma unroll
    for (int r = 0; r < 8; ++r) {
      float v = acc[r] + bv;
      if (pr)     v += pr[(long)r * g.ldr];
      if (g.relu) v = fmaxf(v, 0.0f);
      if (pf)     pf[(long)r * g.ldcf] = v;
      if (pb)     pb[g.transOut ? r : (long)r * g.ldcb] = f2bf(v);
    }
  }
}

// ---------------------------------------------------------------------------
// Transpose + convert: out[n][k] = bf16(in[k][n]); 32x32 LDS tiles,
// batched via blockIdx.z. Used once per layer to stage weights as [N,K] bf16.
// ---------------------------------------------------------------------------
__global__ __launch_bounds__(256) void transpose_cvt_kernel(
    const float* __restrict__ in, __bf16* __restrict__ out, int K, int N) {
  __shared__ float tile[32][33];
  in  += (long)blockIdx.z * K * N;
  out += (long)blockIdx.z * K * N;
  const int kt = blockIdx.x * 32, nt = blockIdx.y * 32;
  const int tx = threadIdx.x & 31, ty = threadIdx.x >> 5;   // 32x8
#pragma unroll
  for (int dy = 0; dy < 32; dy += 8) {
    const int k = kt + ty + dy, n = nt + tx;
    tile[ty + dy][tx] = (k < K && n < N) ? in[(long)k * N + n] : 0.0f;
  }
  __syncthreads();
#pragma unroll
  for (int dy = 0; dy < 32; dy += 8) {
    const int n = nt + ty + dy, k = kt + tx;
    if (n < N && k < K) out[(long)n * K + k] = f2bf(tile[tx][ty + dy]);
  }
}

// ---------------------------------------------------------------------------
// Conv embedder: one block per patch; 10-layer conv stack in LDS, fused
// location embedding. Writes hp (f32) and hb (bf16) rows 0..99.
// ---------------------------------------------------------------------------
struct ConvPtrs { const float* w[10]; const float* b[10]; };

__global__ __launch_bounds__(256) void conv_embed_kernel(
    const float* __restrict__ x, ConvPtrs cp,
    float* __restrict__ hp, __bf16* __restrict__ hb) {
  __shared__ float buf0[6400];
  __shared__ float buf1[6400];

  const int n = blockIdx.x;            // patch index = i*10 + j
  const int i = n / 10, j = n % 10;
  const int tid = threadIdx.x;

  for (int idx = tid; idx < 3 * 20 * 20; idx += blockDim.x) {
    const int c = idx / 400, rem = idx % 400, y = rem / 20, xx = rem % 20;
    buf0[idx] = x[c * 40000 + (20 * j + y) * 200 + (20 * i + xx)];
  }
  __syncthreads();

  const int  OC[10] = {8, 16, 32, 32, 64, 64, 128, 128, 256, 512};
  const int  IC[10] = {3, 8, 16, 32, 32, 64, 64, 128, 128, 256};
  const int  KS[10] = {3, 3, 3, 3, 3, 3, 3, 3, 3, 2};
  const int  IW[10] = {20, 18, 16, 14, 12, 10, 8, 6, 4, 2};
  const bool RL[10] = {false, true, false, true, true, true, true, true, true, false};

  float* in  = buf0;
  float* out = buf1;
  for (int L = 0; L < 10; ++L) {
    const int oc = OC[L], ic = IC[L], ks = KS[L], iw = IW[L];
    const int ow = iw - ks + 1;
    const float* w  = cp.w[L];
    const float* bb = cp.b[L];
    const int total = oc * ow * ow;
    for (int idx = tid; idx < total; idx += blockDim.x) {
      const int o = idx / (ow * ow), rem = idx % (ow * ow);
      const int oy = rem / ow, ox = rem % ow;
      float acc = bb[o];
      for (int c = 0; c < ic; ++c)
        for (int ky = 0; ky < ks; ++ky)
          for (int kx = 0; kx < ks; ++kx)
            acc += in[(c * iw + oy + ky) * iw + ox + kx] *
                   w[((o * ic + c) * ks + ky) * ks + kx];
      if (RL[L]) acc = fmaxf(acc, 0.0f);
      out[idx] = acc;
    }
    __syncthreads();
    float* tmp = in; in = out; out = tmp;
  }

  const float pos = 20.0f * (float)i;
  const float LOG2_10000 = 13.287712379549449f;
  for (int d = tid; d < 512; d += blockDim.x) {
    const int kidx = (d < 256) ? d : d - 256;
    const float ang = pos * exp2f(-(2.0f * (float)kidx / 256.0f) * LOG2_10000);
    const float e = (d < 256) ? sinf(ang) : cosf(ang);
    const float v = in[d] + e;
    hp[n * 512 + d] = v;
    hb[n * 512 + d] = f2bf(v);
  }
}

// ---------------------------------------------------------------------------
// Softmax over padded f32 scores Sf[4][112][128] -> bf16 probs Ab (same
// shape). Valid rows: softmax(0.1*S[0:100]) in cols 0..99, zeros elsewhere.
// Pad rows: all zero.
// ---------------------------------------------------------------------------
__global__ __launch_bounds__(128) void softmax_kernel(
    const float* __restrict__ Sf, __bf16* __restrict__ Ab) {
  __shared__ float red[128];
  const int row = blockIdx.x;           // 0..447, m = row % 112
  const int m   = row % 112;
  const float* p = Sf + (long)row * 128;
  __bf16*      q = Ab + (long)row * 128;
  const int tid = threadIdx.x;
  const __bf16 z = __builtin_bit_cast(__bf16, (unsigned short)0);
  const float scale = 0.1f;             // 1/sqrt(100), faithful quirk

  if (m >= 100) { q[tid] = z; return; }

  float mx = -3.4e38f;
  if (tid < 100) mx = p[tid] * scale;
  red[tid] = mx; __syncthreads();
  for (int s = 64; s > 0; s >>= 1) {
    if (tid < s) red[tid] = fmaxf(red[tid], red[tid + s]);
    __syncthreads();
  }
  mx = red[0]; __syncthreads();

  float e = 0.0f;
  if (tid < 100) e = expf(p[tid] * scale - mx);
  red[tid] = e; __syncthreads();
  for (int s = 64; s > 0; s >>= 1) {
    if (tid < s) red[tid] += red[tid + s];
    __syncthreads();
  }
  const float inv = 1.0f / red[0];
  q[tid] = (tid < 100) ? f2bf(e * inv) : z;
}

__global__ void zero_f32_kernel(float* p, int n) {
  const int i = blockIdx.x * blockDim.x + threadIdx.x;
  if (i < n) p[i] = 0.0f;
}
__global__ void zero_bf16_kernel(__bf16* p, int n) {
  const int i = blockIdx.x * blockDim.x + threadIdx.x;
  if (i < n) p[i] = __builtin_bit_cast(__bf16, (unsigned short)0);
}

// ---------------------------------------------------------------------------
// Host orchestration
// ---------------------------------------------------------------------------
static void launch_gemm(const __bf16* A, const __bf16* B,
                        float* Cf, __bf16* Cb,
                        const float* bias, const float* residual,
                        int M, int N, int K,
                        int lda, int ldb, int ldcf, int ldcb, int ldr,
                        long sA, long sB, long sCf, long sCb, long sR,
                        int batch, int relu, int transOut, hipStream_t stream) {
  GemmArgs g;
  g.A = A; g.B = B; g.Cf = Cf; g.Cb = Cb; g.bias = bias; g.residual = residual;
  g.M = M; g.N = N; g.K = K;
  g.lda = lda; g.ldb = ldb; g.ldcf = ldcf; g.ldcb = ldcb; g.ldr = ldr;
  g.sA = sA; g.sB = sB; g.sCf = sCf; g.sCb = sCb; g.sR = sR;
  g.batch = batch; g.relu = relu; g.transOut = transOut;
  const long tiles = (long)(M / 16) * (N / 16) * batch;
  const int blocks = (int)((tiles + 7) / 8);
  wmma_gemm_kernel<<<blocks, 256, 0, stream>>>(g);
}

static void launch_tr(const float* in, __bf16* out, int K, int N, int batch,
                      hipStream_t stream) {
  dim3 grid((K + 31) / 32, (N + 31) / 32, batch);
  transpose_cvt_kernel<<<grid, 256, 0, stream>>>(in, out, K, N);
}

extern "C" void kernel_launch(void* const* d_in, const int* in_sizes, int n_in,
                              void* d_out, int out_size, void* d_ws, size_t ws_size,
                              hipStream_t stream) {
  (void)in_sizes; (void)n_in; (void)out_size; (void)ws_size;

  const int N_LOC = 100, MP = 112;           // tokens, padded to 7*16
  const int D = 512, H = 4, DK = 128, DV = 256, DFF = 2048;
  const int SP = 128;                        // padded key/score width (4*32)
  const int N_LAYERS = 12;

  const float* x = (const float*)d_in[0];
  ConvPtrs cp;
  for (int i = 0; i < 10; ++i) {
    cp.w[i] = (const float*)d_in[1 + 2 * i];
    cp.b[i] = (const float*)d_in[2 + 2 * i];
  }
  const float* Wq = (const float*)d_in[21];
  const float* Wk = (const float*)d_in[22];
  const float* Wv = (const float*)d_in[23];
  const float* Wo = (const float*)d_in[24];
  const float* W1 = (const float*)d_in[25];
  const float* b1 = (const float*)d_in[26];
  const float* W2 = (const float*)d_in[27];
  const float* b2 = (const float*)d_in[28];

  // ---- workspace layout ----
  float* hp = (float*)d_ws;                  // f32 [112][512] hidden (padded)
  float* Sf = hp + MP * D;                   // f32 [4][112][128] scores
  __bf16* hb  = (__bf16*)(Sf + H * MP * SP); // bf16 [112][512]
  __bf16* Qb  = hb  + MP * D;                // [4][112][128]
  __bf16* Kb  = Qb  + H * MP * DK;           // [4][112][128]
  __bf16* Vt  = Kb  + H * MP * DK;           // [4][256][128]  V^T (padded keys)
  __bf16* Ab  = Vt  + H * DV * SP;           // [4][112][128]  attn probs
  __bf16* Ob  = Ab  + H * MP * SP;           // [112][1024]
  __bf16* Tb  = Ob  + MP * H * DV;           // [112][2048]
  __bf16* wqT = Tb  + MP * DFF;              // [4][128][512]  staging (reused)
  __bf16* wkT = wqT + H * DK * D;            // [4][128][512]
  __bf16* wvT = wkT + H * DK * D;            // [4][256][512]
  __bf16* woT = wvT + H * DV * D;            // [512][1024]
  __bf16* w1T = woT + D * H * DV;            // [2048][512]
  __bf16* w2T = w1T + DFF * D;               // [512][2048]

  // ---- zero pads (once per call) ----
  zero_f32_kernel<<<(12 * D + 255) / 256, 256, 0, stream>>>(hp + N_LOC * D, 12 * D);
  zero_bf16_kernel<<<(12 * D + 255) / 256, 256, 0, stream>>>(hb + N_LOC * D, 12 * D);
  zero_bf16_kernel<<<(H * DV * SP + 255) / 256, 256, 0, stream>>>(Vt, H * DV * SP);

  // ---- conv stack + location embedding -> hp / hb rows 0..99 ----
  conv_embed_kernel<<<N_LOC, 256, 0, stream>>>(x, cp, hp, hb);

  // ---- 12 encoder layers ----
  for (int L = 0; L < N_LAYERS; ++L) {
    const float* bb1 = b1 + (long)L * DFF;
    const float* bb2 = b2 + (long)L * D;

    // Stage this layer's weights as bf16 [N,K]
    launch_tr(Wq + (long)L * H * D * DK, wqT, D, DK, H, stream);
    launch_tr(Wk + (long)L * H * D * DK, wkT, D, DK, H, stream);
    launch_tr(Wv + (long)L * H * D * DV, wvT, D, DV, H, stream);
    launch_tr(Wo + (long)L * (H * DV) * D, woT, H * DV, D, 1, stream);
    launch_tr(W1 + (long)L * D * DFF, w1T, D, DFF, 1, stream);
    launch_tr(W2 + (long)L * DFF * D, w2T, DFF, D, 1, stream);

    // Q/K projections -> bf16 [4][112][128]
    launch_gemm(hb, wqT, nullptr, Qb, nullptr, nullptr, MP, DK, D,
                D, D, 0, DK, 0,
                0, (long)DK * D, 0, (long)MP * DK, 0, H, 0, 0, stream);
    launch_gemm(hb, wkT, nullptr, Kb, nullptr, nullptr, MP, DK, D,
                D, D, 0, DK, 0,
                0, (long)DK * D, 0, (long)MP * DK, 0, H, 0, 0, stream);
    // V projection -> transposed bf16 V^T [4][256][128] (pad keys stay 0)
    launch_gemm(hb, wvT, nullptr, Vt, nullptr, nullptr, MP, DV, D,
                D, D, 0, SP, 0,
                0, (long)DV * D, 0, (long)DV * SP, 0, H, 0, 1, stream);

    // S = Q @ K^T -> f32 scores
    launch_gemm(Qb, Kb, Sf, nullptr, nullptr, nullptr, MP, MP, DK,
                DK, DK, SP, 0, 0,
                (long)MP * DK, (long)MP * DK, (long)MP * SP, 0, 0,
                H, 0, 0, stream);

    // softmax (scale 0.1) -> bf16 probs, zero pads
    softmax_kernel<<<H * MP, 128, 0, stream>>>(Sf, Ab);

    // O = A @ V  (B = V^T, NT) -> interleaved bf16 [112][1024]
    launch_gemm(Ab, Vt, nullptr, Ob, nullptr, nullptr, MP, DV, SP,
                SP, SP, 0, H * DV, 0,
                (long)MP * SP, (long)DV * SP, 0, (long)DV, 0,
                H, 0, 0, stream);

    // h = h + O @ Wo  -> hp (f32) + hb (bf16)
    launch_gemm(Ob, woT, hp, hb, nullptr, hp, MP, D, H * DV,
                H * DV, H * DV, D, D, D,
                0, 0, 0, 0, 0, 1, 0, 0, stream);

    // T = relu(h @ W1 + b1) -> bf16 [112][2048]
    launch_gemm(hb, w1T, nullptr, Tb, bb1, nullptr, MP, DFF, D,
                D, D, 0, DFF, 0,
                0, 0, 0, 0, 0, 1, 1, 0, stream);

    // h = h + T @ W2 + b2 -> hp (f32) + hb (bf16)
    launch_gemm(Tb, w2T, hp, hb, bb2, hp, MP, D, DFF,
                DFF, DFF, D, D, D,
                0, 0, 0, 0, 0, 1, 0, 0, stream);
  }

  // Export the 100 valid rows (contiguous) to d_out.
  hipMemcpyAsync(d_out, hp, (size_t)N_LOC * D * sizeof(float),
                 hipMemcpyDeviceToDevice, stream);
}
